// LSTM_32564442038642
// MI455X (gfx1250) — matrix operand order repaired and checked by reading, not compiled, not run on hardware
//
#include <hip/hip_runtime.h>

// LSTM: B=128, S=512, I=256, H=256, O=128.
//   - 8 workgroups, each owns a 16-row batch tile for all 512 timesteps (batch
//     rows are recurrence-independent -> no cross-WGP sync ever).
//   - 16 waves/WG; wave w owns hidden units [16w,16w+16) and computes its f,i,g,o
//     gate tiles with v_wmma_f32_16x16x32_bf16 (f32 accum), cell state in VGPRs.
//   - Weights pre-packed to per-lane WMMA B-fragment order, streamed from L2
//     (1MB bf16 resident in 192MB L2 across all 512 steps).
//   - x pre-converted to bf16 once; per-step x_t tile staged into double-buffered
//     LDS via GLOBAL_LOAD_ASYNC_TO_LDS_B128 one step ahead (ASYNCcnt), fully
//     overlapped with the GEMM of the previous step.
//   - K loop kept as two partially-unrolled real loops (x half / h half) so the
//     t-invariant weight loads are NOT hoisted+spilled by LICM (seen in round 3).

#define SEQ   512
#define DIN   256
#define HID   256
#define DD    512
#define OUTD  128
#define XW    264          // padded LDS row stride (bf16 elems): 132 dwords -> +4 banks/row

typedef __attribute__((ext_vector_type(16))) __bf16 v16bf;
typedef __attribute__((ext_vector_type(8)))  float  v8f;
typedef __attribute__((ext_vector_type(4)))  int    v4i_;
typedef __attribute__((address_space(1))) v4i_* gptr128;   // global int4*
typedef __attribute__((address_space(3))) v4i_* lptr128;   // LDS int4*

struct Frag32 { uint4 lo, hi; };   // 32 bytes == one v16bf fragment

__device__ __forceinline__ unsigned short f2bf(float f) {
  union { float f; unsigned u; } v; v.f = f;
  unsigned r = v.u + 0x7FFFu + ((v.u >> 16) & 1u);   // round-to-nearest-even
  return (unsigned short)(r >> 16);
}

__device__ __forceinline__ float sigm(float x) {
  return 1.0f / (1.0f + __expf(-x));
}

// ---- async global->LDS b128 (ASYNCcnt path)
__device__ __forceinline__ void async_load_b128(const void* g, unsigned lds_off) {
#if __has_builtin(__builtin_amdgcn_global_load_async_to_lds_b128)
  __builtin_amdgcn_global_load_async_to_lds_b128(
      (gptr128)(unsigned long long)(size_t)g,
      (lptr128)(unsigned long long)lds_off, 0, 0);
#else
  asm volatile("global_load_async_to_lds_b128 %0, %1, off"
               :: "v"(lds_off), "v"((unsigned long long)(size_t)g) : "memory");
#endif
}

template <int N>
__device__ __forceinline__ void wait_async() {
#if __has_builtin(__builtin_amdgcn_s_wait_asynccnt)
  __builtin_amdgcn_s_wait_asynccnt(N);
#else
  asm volatile("s_wait_asynccnt %0" :: "i"(N) : "memory");
#endif
}

// ---- Pre-pack Wg = concat(Wf,Wi,Wc,Wo) [1024 x 512] -> bf16, WMMA B-fragment order.
// dst[((n_t*16 + kc)*32 + lane)*16 + j] = Wgate[g][(w*16 + (lane&15))*512 + kc*32 + ((j&8)<<1) + 8*(lane>>4) + (j&7)],
//   g = n_t>>4, w = n_t&15  (wave w's 4 gate tiles are n_t = g*16 + w).
__global__ void pack_gates_kernel(const float* __restrict__ Wf, const float* __restrict__ Wi,
                                  const float* __restrict__ Wc, const float* __restrict__ Wo,
                                  unsigned short* __restrict__ dst) {
  int idx  = blockIdx.x * blockDim.x + threadIdx.x;   // [0, 524288)
  int j    = idx & 15;
  int lane = (idx >> 4) & 31;
  int kc   = (idx >> 9) & 15;
  int n_t  = idx >> 13;
  int g = n_t >> 4, w = n_t & 15;
  const float* W = (g == 0) ? Wf : (g == 1) ? Wi : (g == 2) ? Wc : Wo;
  dst[idx] = f2bf(W[(w * 16 + (lane & 15)) * DD + kc * 32 + ((j & 8) << 1) + 8 * (lane >> 4) + (j & 7)]);
}

// ---- Pre-pack Wout [128 x 256] the same way: 8 n-tiles x 8 k-chunks.
__global__ void pack_wout_kernel(const float* __restrict__ Wout,
                                 unsigned short* __restrict__ dst) {
  int idx  = blockIdx.x * blockDim.x + threadIdx.x;   // [0, 32768)
  int j    = idx & 15;
  int lane = (idx >> 4) & 31;
  int kc   = (idx >> 9) & 7;
  int n_t  = idx >> 12;
  dst[idx] = f2bf(Wout[(n_t * 16 + (lane & 15)) * HID + kc * 32 + ((j & 8) << 1) + 8 * (lane >> 4) + (j & 7)]);
}

// ---- Pre-convert x (f32 -> bf16), 4 elems/thread.
__global__ void convert_x_kernel(const float* __restrict__ x,
                                 unsigned short* __restrict__ dst) {
  int i = blockIdx.x * blockDim.x + threadIdx.x;      // [0, 4194304)
  float4 v = ((const float4*)x)[i];
  uint2 o;
  o.x = (unsigned)f2bf(v.x) | ((unsigned)f2bf(v.y) << 16);
  o.y = (unsigned)f2bf(v.z) | ((unsigned)f2bf(v.w) << 16);
  ((uint2*)dst)[i] = o;
}

template <bool ASYNCX>
__global__ void __launch_bounds__(512)
lstm_kernel(const float* __restrict__ xf32,             // used when !ASYNCX
            const unsigned short* __restrict__ xbf,     // used when ASYNCX
            const float* __restrict__ bf_, const float* __restrict__ bi_,
            const float* __restrict__ bc_, const float* __restrict__ bo_,
            const float* __restrict__ bout,
            const unsigned short* __restrict__ Wg,      // packed gate weights (1MB)
            const unsigned short* __restrict__ W2,      // packed Wout (64KB)
            float* __restrict__ out) {
  __shared__ unsigned short xbuf[(ASYNCX ? 2 : 1) * 16 * XW];  // x_t tile(s), bf16
  __shared__ unsigned short hbuf[16 * XW];                     // h tile, bf16

  const int tid  = threadIdx.x;
  const int lane = tid & 31;
  const int w    = tid >> 5;        // wave id 0..15 (uniform per wave)
  const int l16  = lane & 15;
  const int hi   = lane >> 4;
  const int n    = w * 16 + l16;    // hidden unit owned by this lane
  const int row0 = blockIdx.x * 16; // batch-row base of this workgroup
  const int m    = tid >> 5;        // staging row
  const int i0   = (tid & 31) * 8;  // staging col

  // zero h (h0 = 0)
  {
    uint4 z = make_uint4(0u, 0u, 0u, 0u);
    *(uint4*)&hbuf[m * XW + i0] = z;
  }

  v8f c;
  #pragma unroll
  for (int r = 0; r < 8; ++r) c[r] = 0.0f;

  const float bias[4] = { bf_[n], bi_[n], bc_[n], bo_[n] };
  const int rowx = l16 * XW;
  // per-wave weight base: fragment for (gate g, chunk kc) at  bbase + g*131072 + kc*512
  const unsigned short* bbase = Wg + (size_t)w * 16 * 512 + lane * 16;

  if (ASYNCX) {   // kick off x(0) into buffer 0
    const unsigned short* g =
        xbf + ((size_t)(row0 + m) * SEQ + 0) * DIN + i0;
    async_load_b128(g, (unsigned)(size_t)&xbuf[m * XW + i0]);
  }

  for (int t = 0; t < SEQ; ++t) {
    const unsigned short* xb;
    if (ASYNCX) {
      xb = &xbuf[(t & 1) * 16 * XW];
      if (t + 1 < SEQ) {   // prefetch x(t+1) into the other buffer, then make
                           // sure the older load (x(t)) has landed (in-order).
        const unsigned short* g =
            xbf + ((size_t)(row0 + m) * SEQ + (t + 1)) * DIN + i0;
        async_load_b128(g, (unsigned)(size_t)&xbuf[((t + 1) & 1) * 16 * XW + m * XW + i0]);
        wait_async<1>();
      } else {
        wait_async<0>();
      }
    } else {
      xb = xbuf;
      const float4* px =
          (const float4*)(xf32 + ((size_t)(row0 + m) * SEQ + t) * DIN + i0);
      float4 a0 = px[0], a1 = px[1];
      uint4 o;
      o.x = (unsigned)f2bf(a0.x) | ((unsigned)f2bf(a0.y) << 16);
      o.y = (unsigned)f2bf(a0.z) | ((unsigned)f2bf(a0.w) << 16);
      o.z = (unsigned)f2bf(a1.x) | ((unsigned)f2bf(a1.y) << 16);
      o.w = (unsigned)f2bf(a1.z) | ((unsigned)f2bf(a1.w) << 16);
      *(uint4*)&xbuf[m * XW + i0] = o;
    }
    __syncthreads();   // x(t) staged everywhere AND h(t-1) writes visible

    // ---- gate GEMM: 4 accumulators (f,i,g,o), K = 512 in 16 chunks of 32.
    v8f acc[4];
    #pragma unroll
    for (int g = 0; g < 4; ++g)
      #pragma unroll
      for (int r = 0; r < 8; ++r) acc[g][r] = bias[g];

    // K in [0,256): A fragments from the x buffer.  Real loop (unroll 4) so the
    // t-invariant weight loads keep their induction and are not hoisted/spilled.
    #pragma unroll 4
    for (int kc = 0; kc < 8; ++kc) {
      const unsigned short* asrc = &xb[rowx + kc * 32 + 8 * hi];
      Frag32 af;
      af.lo = *(const uint4*)asrc;
      af.hi = *(const uint4*)(asrc + 16);
      v16bf a = __builtin_bit_cast(v16bf, af);
      __builtin_prefetch(bbase + (kc + 1) * 512, 0, 3);
      #pragma unroll
      for (int g = 0; g < 4; ++g) {
        const unsigned short* bp = bbase + g * 131072 + kc * 512;
        Frag32 bfr;
        bfr.lo = *(const uint4*)bp;
        bfr.hi = *(const uint4*)(bp + 8);
        v16bf b = __builtin_bit_cast(v16bf, bfr);
        acc[g] = __builtin_amdgcn_wmma_f32_16x16x32_bf16(
            false, a, false, b, (short)0, acc[g], false, false);
      }
    }
    // K in [256,512): A fragments from the h buffer.
    #pragma unroll 4
    for (int kc = 8; kc < 16; ++kc) {
      const unsigned short* asrc = &hbuf[rowx + (kc - 8) * 32 + 8 * hi];
      Frag32 af;
      af.lo = *(const uint4*)asrc;
      af.hi = *(const uint4*)(asrc + 16);
      v16bf a = __builtin_bit_cast(v16bf, af);
      #pragma unroll
      for (int g = 0; g < 4; ++g) {
        const unsigned short* bp = bbase + g * 131072 + kc * 512;
        Frag32 bfr;
        bfr.lo = *(const uint4*)bp;
        bfr.hi = *(const uint4*)(bp + 8);
        v16bf b = __builtin_bit_cast(v16bf, bfr);
        acc[g] = __builtin_amdgcn_wmma_f32_16x16x32_bf16(
            false, a, false, b, (short)0, acc[g], false, false);
      }
    }
    __syncthreads();   // all waves done reading hbuf before it is overwritten

    // ---- cell update in registers; write new h (bf16) into hbuf
    #pragma unroll
    for (int r = 0; r < 8; ++r) {
      float fg = sigm(acc[0][r]);
      float ig = sigm(acc[1][r]);
      float gg = tanhf(acc[2][r]);
      float og = sigm(acc[3][r]);
      float cn = c[r] * fg + ig * gg;
      c[r] = cn;
      float hn = tanhf(cn) * og;
      hbuf[(r + 8 * hi) * XW + n] = f2bf(hn);
    }
  }
  __syncthreads();     // final h visible

  // ---- out = h_final @ Wout^T + bout : waves 0..7, one 16x16 out tile each
  if (w < 8) {
    v8f acc;
    float bb = bout[w * 16 + l16];
    #pragma unroll
    for (int r = 0; r < 8; ++r) acc[r] = bb;
    #pragma unroll 4
    for (int kc = 0; kc < 8; ++kc) {
      const unsigned short* asrc = &hbuf[rowx + kc * 32 + 8 * hi];
      Frag32 af;
      af.lo = *(const uint4*)asrc;
      af.hi = *(const uint4*)(asrc + 16);
      v16bf a = __builtin_bit_cast(v16bf, af);
      const unsigned short* bp = W2 + (size_t)(w * 8 + kc) * 512 + lane * 16;
      Frag32 bfr;
      bfr.lo = *(const uint4*)bp;
      bfr.hi = *(const uint4*)(bp + 8);
      v16bf b = __builtin_bit_cast(v16bf, bfr);
      acc = __builtin_amdgcn_wmma_f32_16x16x32_bf16(
          false, a, false, b, (short)0, acc, false, false);
    }
    #pragma unroll
    for (int r = 0; r < 8; ++r)
      out[(size_t)(row0 + r + 8 * hi) * OUTD + w * 16 + l16] = acc[r];
  }
}

extern "C" void kernel_launch(void* const* d_in, const int* in_sizes, int n_in,
                              void* d_out, int out_size, void* d_ws, size_t ws_size,
                              hipStream_t stream) {
  (void)in_sizes; (void)n_in; (void)out_size;
  const float* x    = (const float*)d_in[0];
  const float* Wf   = (const float*)d_in[1];
  const float* bf_  = (const float*)d_in[2];
  const float* Wi   = (const float*)d_in[3];
  const float* bi_  = (const float*)d_in[4];
  const float* Wc   = (const float*)d_in[5];
  const float* bc_  = (const float*)d_in[6];
  const float* Wo   = (const float*)d_in[7];
  const float* bo_  = (const float*)d_in[8];
  const float* Wout = (const float*)d_in[9];
  const float* bout = (const float*)d_in[10];

  unsigned short* ws        = (unsigned short*)d_ws;
  unsigned short* Wg_packed = ws;                       // 524288 bf16 = 1 MB
  unsigned short* W2_packed = ws + 524288;              // 32768  bf16 = 64 KB
  unsigned short* x_bf16    = ws + 524288 + 32768;      // 16777216 bf16 = 32 MB

  const size_t need_async = (size_t)(524288 + 32768 + 16777216) * 2;

  pack_gates_kernel<<<1024, 512, 0, stream>>>(Wf, Wi, Wc, Wo, Wg_packed);
  pack_wout_kernel<<<64, 512, 0, stream>>>(Wout, W2_packed);

  if (ws_size >= need_async) {
    convert_x_kernel<<<8192, 512, 0, stream>>>(x, x_bf16);
    lstm_kernel<true><<<8, 512, 0, stream>>>(x, x_bf16, bf_, bi_, bc_, bo_, bout,
                                             Wg_packed, W2_packed, (float*)d_out);
  } else {
    lstm_kernel<false><<<8, 512, 0, stream>>>(x, x_bf16, bf_, bi_, bc_, bo_, bout,
                                              Wg_packed, W2_packed, (float*)d_out);
  }
}
